// IntraGraphAttention_4483945857420
// MI455X (gfx1250) — compile-verified, matrix operand order checked by reading, not run on hardware
//
#include <hip/hip_runtime.h>
#include <cstdint>
#include <cstddef>

#define NNODES 50000
#define NEDGES 1250000
#define IND 128
#define OUTD 64
#define CH 32
#define NEG_SLOPE 0.2f

typedef __attribute__((ext_vector_type(16))) _Float16 v16h;
typedef __attribute__((ext_vector_type(8)))  float    v8f;

// ---- workspace layout (offsets in floats) ----
#define OFF_H      0ull        // [N*64]   h = relu(x)@W
#define OFF_AED    3200000ull  // [E*2]    per-edge a_edge
#define OFF_ALPHA  5700000ull  // [E*2]    logits -> exp(p) in place
#define OFF_ASRC   8200000ull  // [N*2]
#define OFF_ADST   8300000ull  // [N*2]
#define OFF_ALOOP  8400000ull  // [N*2]    self-loop logits -> exp(p)
#define OFF_V      8500000ull  // [64*2]   W_edge @ att_edge (per head)
// contiguous zero-block below (memset each launch)
#define OFF_DEG    8500224ull  // [N]      incoming-edge count (float)
#define OFF_LOOP   8550224ull  // [N*2]    segment_sum(a_edge, dst)
#define OFF_AMAX   8650224ull  // [N*2]    encoded segment max (uint)
#define OFF_DENOM  8750224ull  // [N*2]    softmax denominator
#define OFF_END    8850224ull
#define ZERO_BYTES ((OFF_END - OFF_DEG) * 4ull)

// monotone float <-> uint encoding for atomicMax-based segment max
__device__ __forceinline__ unsigned encf(float f) {
  unsigned u = __float_as_uint(f);
  return (u & 0x80000000u) ? ~u : (u | 0x80000000u);
}
__device__ __forceinline__ float decf(unsigned u) {
  unsigned b = (u & 0x80000000u) ? (u & 0x7FFFFFFFu) : ~u;
  return __uint_as_float(b);
}

// ---- K0: V[d][h] = sum_c W_edge[d, h*32+c] * att_edge[h,c] ----
__global__ void gat_edge_vec(const float* __restrict__ W_edge,
                             const float* __restrict__ att_edge,
                             float* __restrict__ V) {
  int d = threadIdx.x;
  if (d >= 64) return;
  float v0 = 0.f, v1 = 0.f;
  for (int c = 0; c < CH; ++c) {
    v0 += W_edge[d * OUTD + c]      * att_edge[c];
    v1 += W_edge[d * OUTD + CH + c] * att_edge[CH + c];
  }
  V[d * 2] = v0; V[d * 2 + 1] = v1;
}

// ---- K1: h = relu(x) @ W via v_wmma_f32_16x16x32_f16 ----
// Block = 256 threads = 8 waves; one wave per 16-node tile (N % 16 == 0, so
// no per-row guards). W staged once in LDS as per-lane f16 B-fragments
// (lane<->N column, 16 halves<->K). D tiles are transposed through a
// wave-private LDS staging buffer so global stores are coalesced b128.
__global__ void __launch_bounds__(256)
gat_h_wmma(const float* __restrict__ x, const float* __restrict__ W,
           float* __restrict__ hbuf) {
  __shared__ __align__(32) _Float16 bfrag[4][4][32][16]; // [kb][nb][lane][j] 16 KB
  __shared__ __align__(16) float    tbuf[8][16][64];     // per-wave 16x64 tile, 32 KB
  const int tid = threadIdx.x;
  for (int i = tid; i < 4 * 4 * 32 * 16; i += 256) {
    int j = i & 15, lane = (i >> 4) & 31, nb = (i >> 9) & 3, kb = (i >> 11) & 3;
    int klocal = (j < 8 ? j : j + 8) + ((lane & 16) >> 1);  // +8 for hi half-wave
    int k = kb * 32 + klocal;
    int n = nb * 16 + (lane & 15);
    bfrag[kb][nb][lane][j] = (_Float16)W[k * OUTD + n];
  }
  __syncthreads();

  const int lane = tid & 31, wave = tid >> 5;
  const int tile = (blockIdx.x * 8 + wave) * 16;
  if (tile >= NNODES) return;           // wave-uniform: EXEC stays all-ones for WMMA

  const int row = tile + (lane & 15);   // always < NNODES (N multiple of 16)
  const float* xr = x + (size_t)row * IND;

  v16h a[4];
#pragma unroll
  for (int kb = 0; kb < 4; ++kb) {
    int b0 = kb * 32 + ((lane & 16) >> 1);
    v16h av;
#pragma unroll
    for (int j = 0; j < 8; ++j) { float t = xr[b0 + j];      av[j]     = (_Float16)(t > 0.f ? t : 0.f); }
#pragma unroll
    for (int j = 0; j < 8; ++j) { float t = xr[b0 + 16 + j]; av[8 + j] = (_Float16)(t > 0.f ? t : 0.f); }
    a[kb] = av;
  }

  const int mbase = (lane & 16) >> 1;   // D: M = vgpr + 8*(lane>=16), N = lane%16
#pragma unroll
  for (int nb = 0; nb < 4; ++nb) {
    v8f c = {};
#pragma unroll
    for (int kb = 0; kb < 4; ++kb) {
      v16h b = *(const v16h*)&bfrag[kb][nb][lane][0];
      c = __builtin_amdgcn_wmma_f32_16x16x32_f16(false, a[kb], false, b,
                                                 (short)0, c, false, false);
    }
    int col = nb * 16 + (lane & 15);
#pragma unroll
    for (int v = 0; v < 8; ++v) tbuf[wave][v + mbase][col] = c[v];
  }

  // wave-private LDS region; DS ops from one wave are in-order -> no barrier.
  // 16x64 tile is contiguous in hbuf: 256 float4s, 8 per lane, fully coalesced.
  const float4* tsrc = (const float4*)&tbuf[wave][0][0];
  float4* dst4 = (float4*)(hbuf + (size_t)tile * OUTD);
#pragma unroll
  for (int i = 0; i < 8; ++i) dst4[i * 32 + lane] = tsrc[i * 32 + lane];
}

// ---- K2: a_src/a_dst per node ----
__global__ void gat_node_att(const float* __restrict__ hbuf,
                             const float* __restrict__ att_src,
                             const float* __restrict__ att_dst,
                             float* __restrict__ asrc, float* __restrict__ adst) {
  int n = blockIdx.x * blockDim.x + threadIdx.x;
  if (n >= NNODES) return;
  const float* hr = hbuf + (size_t)n * OUTD;
  float s0 = 0.f, s1 = 0.f, d0 = 0.f, d1 = 0.f;
  for (int c = 0; c < CH; ++c) {
    float h0 = hr[c], h1 = hr[CH + c];
    s0 += h0 * att_src[c];  s1 += h1 * att_src[CH + c];
    d0 += h0 * att_dst[c];  d1 += h1 * att_dst[CH + c];
  }
  asrc[n * 2] = s0; asrc[n * 2 + 1] = s1;
  adst[n * 2] = d0; adst[n * 2 + 1] = d1;
}

// ---- K3: per-edge a_edge (matvec vs V) + deg + loop sums. 4 threads/edge ----
__global__ void gat_aedge(const float* __restrict__ eattr, const float* __restrict__ V,
                          const int* __restrict__ dst, float* __restrict__ aed,
                          float* __restrict__ deg, float* __restrict__ loop_aed) {
  int t = blockIdx.x * blockDim.x + threadIdx.x;
  int e = t >> 2, q = t & 3;
  if (e >= NEDGES) return;
  const float* ar = eattr + (size_t)e * 64 + q * 16;
  float p0 = 0.f, p1 = 0.f;
#pragma unroll
  for (int i = 0; i < 16; ++i) {
    float a = ar[i];
    int d2 = (q * 16 + i) * 2;
    p0 += a * V[d2]; p1 += a * V[d2 + 1];
  }
  p0 += __shfl_xor(p0, 1, 4); p0 += __shfl_xor(p0, 2, 4);
  p1 += __shfl_xor(p1, 1, 4); p1 += __shfl_xor(p1, 2, 4);
  if (q == 0) {
    aed[e * 2] = p0; aed[e * 2 + 1] = p1;
    int d = dst[e];
    atomicAdd(&deg[d], 1.0f);
    atomicAdd(&loop_aed[d * 2], p0);
    atomicAdd(&loop_aed[d * 2 + 1], p1);
  }
}

// ---- K4: per-edge logits + segment max ----
__global__ void gat_logits(const int* __restrict__ src, const int* __restrict__ dst,
                           const float* __restrict__ asrc, const float* __restrict__ adst,
                           const float* __restrict__ aed, float* __restrict__ alpha,
                           unsigned* __restrict__ amax) {
  int e = blockIdx.x * blockDim.x + threadIdx.x;
  if (e >= NEDGES) return;
  int s = src[e], d = dst[e];
#pragma unroll
  for (int h = 0; h < 2; ++h) {
    float t = asrc[s * 2 + h] + adst[d * 2 + h] + aed[e * 2 + h];
    t = t > 0.f ? t : NEG_SLOPE * t;
    alpha[e * 2 + h] = t;
    atomicMax(&amax[d * 2 + h], encf(t));
  }
}

// ---- K4b: self-loop logits (loop_attr folded: loop_aed/max(deg,1)) ----
__global__ void gat_logits_self(const float* __restrict__ asrc, const float* __restrict__ adst,
                                const float* __restrict__ loop_aed, const float* __restrict__ deg,
                                float* __restrict__ alpha_loop, unsigned* __restrict__ amax) {
  int n = blockIdx.x * blockDim.x + threadIdx.x;
  if (n >= NNODES) return;
  float dg = deg[n]; dg = dg > 1.f ? dg : 1.f;
#pragma unroll
  for (int h = 0; h < 2; ++h) {
    float t = asrc[n * 2 + h] + adst[n * 2 + h] + loop_aed[n * 2 + h] / dg;
    t = t > 0.f ? t : NEG_SLOPE * t;
    alpha_loop[n * 2 + h] = t;
    atomicMax(&amax[n * 2 + h], encf(t));
  }
}

// ---- K5 / K5b: exp(logit - max), accumulate denom ----
__global__ void gat_exp(const int* __restrict__ dst, const unsigned* __restrict__ amax,
                        float* __restrict__ alpha, float* __restrict__ denom) {
  int e = blockIdx.x * blockDim.x + threadIdx.x;
  if (e >= NEDGES) return;
  int d = dst[e];
#pragma unroll
  for (int h = 0; h < 2; ++h) {
    float p = __expf(alpha[e * 2 + h] - decf(amax[d * 2 + h]));
    alpha[e * 2 + h] = p;
    atomicAdd(&denom[d * 2 + h], p);
  }
}

__global__ void gat_exp_self(const unsigned* __restrict__ amax,
                             float* __restrict__ alpha_loop, float* __restrict__ denom) {
  int n = blockIdx.x * blockDim.x + threadIdx.x;
  if (n >= NNODES) return;
#pragma unroll
  for (int h = 0; h < 2; ++h) {
    float p = __expf(alpha_loop[n * 2 + h] - decf(amax[n * 2 + h]));
    alpha_loop[n * 2 + h] = p;
    atomicAdd(&denom[n * 2 + h], p);
  }
}

// ---- K6: scatter-aggregate messages, 16 threads/edge (float4 per thread) ----
__global__ void gat_agg(const int* __restrict__ src, const int* __restrict__ dst,
                        const float* __restrict__ hbuf, const float* __restrict__ alpha,
                        const float* __restrict__ denom, float* __restrict__ out) {
  int t = blockIdx.x * blockDim.x + threadIdx.x;
  int e = t >> 4, part = t & 15;
  if (e >= NEDGES) return;
  int s = src[e], d = dst[e];
  int head = part >> 3;
  float w = alpha[e * 2 + head] / (denom[d * 2 + head] + 1e-16f);
  int ch = part * 4;
  const float4 hv = *(const float4*)(hbuf + (size_t)s * OUTD + ch);
  float* op = out + (size_t)d * OUTD + ch;
  atomicAdd(op + 0, hv.x * w); atomicAdd(op + 1, hv.y * w);
  atomicAdd(op + 2, hv.z * w); atomicAdd(op + 3, hv.w * w);
}

// ---- K6b: self-loop messages + bias ----
__global__ void gat_agg_self(const float* __restrict__ hbuf, const float* __restrict__ alpha_loop,
                             const float* __restrict__ denom, const float* __restrict__ bias,
                             float* __restrict__ out) {
  int t = blockIdx.x * blockDim.x + threadIdx.x;
  int n = t >> 4, part = t & 15;
  if (n >= NNODES) return;
  int head = part >> 3;
  float w = alpha_loop[n * 2 + head] / (denom[n * 2 + head] + 1e-16f);
  int ch = part * 4;
  const float4 hv = *(const float4*)(hbuf + (size_t)n * OUTD + ch);
  float* op = out + (size_t)n * OUTD + ch;
  atomicAdd(op + 0, hv.x * w + bias[ch + 0]);
  atomicAdd(op + 1, hv.y * w + bias[ch + 1]);
  atomicAdd(op + 2, hv.z * w + bias[ch + 2]);
  atomicAdd(op + 3, hv.w * w + bias[ch + 3]);
}

extern "C" void kernel_launch(void* const* d_in, const int* in_sizes, int n_in,
                              void* d_out, int out_size, void* d_ws, size_t ws_size,
                              hipStream_t stream) {
  const float* x        = (const float*)d_in[0];
  const int*   ei       = (const int*)d_in[1];
  const float* eattr    = (const float*)d_in[2];
  const float* W        = (const float*)d_in[3];
  const float* W_edge   = (const float*)d_in[4];
  const float* att_src  = (const float*)d_in[5];
  const float* att_dst  = (const float*)d_in[6];
  const float* att_edge = (const float*)d_in[7];
  const float* bias     = (const float*)d_in[8];
  float* out = (float*)d_out;

  float* ws = (float*)d_ws;
  float*    hbuf   = ws + OFF_H;
  float*    aed    = ws + OFF_AED;
  float*    alpha  = ws + OFF_ALPHA;
  float*    asrc   = ws + OFF_ASRC;
  float*    adst   = ws + OFF_ADST;
  float*    aloop  = ws + OFF_ALOOP;
  float*    V      = ws + OFF_V;
  float*    deg    = ws + OFF_DEG;
  float*    lp     = ws + OFF_LOOP;
  unsigned* amax   = (unsigned*)(ws + OFF_AMAX);
  float*    denom  = ws + OFF_DENOM;

  const int* srcI = ei;           // edge_index[0]
  const int* dstI = ei + NEDGES;  // edge_index[1]

  // re-zero accumulators every call (deterministic across graph replays)
  hipMemsetAsync(ws + OFF_DEG, 0, ZERO_BYTES, stream);
  hipMemsetAsync(out, 0, (size_t)NNODES * OUTD * sizeof(float), stream);

  gat_edge_vec<<<1, 64, 0, stream>>>(W_edge, att_edge, V);

  const int tiles = (NNODES + 15) / 16;                 // 3125
  gat_h_wmma<<<(tiles + 7) / 8, 256, 0, stream>>>(x, W, hbuf);

  gat_node_att<<<(NNODES + 255) / 256, 256, 0, stream>>>(hbuf, att_src, att_dst, asrc, adst);

  gat_aedge<<<((size_t)NEDGES * 4 + 255) / 256, 256, 0, stream>>>(eattr, V, dstI, aed, deg, lp);

  gat_logits<<<(NEDGES + 255) / 256, 256, 0, stream>>>(srcI, dstI, asrc, adst, aed, alpha, amax);
  gat_logits_self<<<(NNODES + 255) / 256, 256, 0, stream>>>(asrc, adst, lp, deg, aloop, amax);

  gat_exp<<<(NEDGES + 255) / 256, 256, 0, stream>>>(dstI, amax, alpha, denom);
  gat_exp_self<<<(NNODES + 255) / 256, 256, 0, stream>>>(amax, aloop, denom);

  gat_agg<<<((size_t)NEDGES * 16) / 256, 256, 0, stream>>>(srcI, dstI, hbuf, alpha, denom, out);
  gat_agg_self<<<((size_t)NNODES * 16) / 256, 256, 0, stream>>>(hbuf, aloop, denom, bias, out);
}
